// STU_64802466562165
// MI455X (gfx1250) — matrix operand order repaired and checked by reading, not compiled, not run on hardware
//
#include <hip/hip_runtime.h>

// ---------------------------------------------------------------------------
// STU forward for MI455X (gfx1250, wave32, WMMA).
//   stage1: conv  = Toeplitz(phi_k * lambda_k^0.25) @ u   (bf16 WMMA, causal,
//                                                          64x64 register tiles)
//   stage2: delta = x_tilde @ m_phi + 3-tap AR(u, m_u)    (bf16 WMMA, shared B
//                                                          panel via async->LDS)
//   stage3: y_t   = [y_{t-1}; y_{t-2}] @ m_y^T + delta_t  (WMMA GEMV steps,
//                                                          LDS double buffer,
//                                                          4-way accum chains)
// ---------------------------------------------------------------------------

#define B_  4
#define L_  1024
#define D_  512
#define K_  24
#define KU_ 3
#define KD_ (K_ * D_)   // 12288

typedef __attribute__((ext_vector_type(16))) __bf16        v16bf;
typedef __attribute__((ext_vector_type(8)))  float         v8f;
typedef __attribute__((ext_vector_type(4)))  unsigned int  v4u;

union Frag32B { v4u q[2]; v16bf bf; };

// f32 -> bf16 round-to-nearest-even
__device__ __forceinline__ unsigned short f2bf(float x) {
  unsigned int u = __float_as_uint(x);
  u += 0x7FFFu + ((u >> 16) & 1u);
  return (unsigned short)(u >> 16);
}

// A-matrix 16x32 bf16 fragment: per lane 32 contiguous bytes (16 K values) at p0
__device__ __forceinline__ v16bf ldA(const unsigned short* p0,
                                     const unsigned short* p1) {
  Frag32B f;
  f.q[0] = *(const v4u*)p0;
  f.q[1] = *(const v4u*)(p0 + 8);
  return f.bf;
}
// B-matrix 32x16 bf16 fragment: per lane 32 contiguous bytes (16 K values)
__device__ __forceinline__ v16bf ldB(const unsigned short* p) {
  Frag32B f;
  f.q[0] = *(const v4u*)p;
  f.q[1] = *(const v4u*)(p + 8);
  return f.bf;
}

__device__ __forceinline__ v8f wmma_bf16(v16bf a, v16bf b, v8f c) {
  // (neg_a, A, neg_b, B, c_mod, C, reuse_a, reuse_b)
  return __builtin_amdgcn_wmma_f32_16x16x32_bf16(false, a, false, b,
                                                 (short)0, c, false, false);
}

// CDNA5 async copy: 16B per lane, global -> LDS, tracked by ASYNCcnt.
// Generic LDS pointer's low 32 bits are the LDS byte offset (ISA aperture rule).
__device__ __forceinline__ void async_ld16(void* lds_ptr, const void* gptr) {
  unsigned int lds_off = (unsigned int)(size_t)lds_ptr;
  asm volatile("global_load_async_to_lds_b128 %0, %1, off"
               :: "v"(lds_off), "v"(gptr)
               : "memory");
}

// ------------------------------ prep kernels -------------------------------

__global__ void prep_inputs(const float* __restrict__ inp,
                            unsigned short* __restrict__ u_bf,   // [B][L][D]
                            unsigned short* __restrict__ u_T) {  // [B][D][L]
  int idx = blockIdx.x * blockDim.x + threadIdx.x;
  if (idx >= B_ * L_ * D_) return;
  int d = idx % D_;
  int t = (idx / D_) % L_;
  int b = idx / (D_ * L_);
  unsigned short v = f2bf(inp[idx]);
  u_bf[idx] = v;
  u_T[((size_t)b * D_ + d) * L_ + t] = v;
}

__global__ void prep_phi(const float* __restrict__ m_phi,          // [KD][D]
                         unsigned short* __restrict__ phiT) {      // [D][KD]
  int idx = blockIdx.x * blockDim.x + threadIdx.x;
  if (idx >= KD_ * D_) return;
  int o = idx % D_;
  int j = idx / D_;
  phiT[(size_t)o * KD_ + j] = f2bf(m_phi[idx]);
}

__global__ void prep_mu(const float* __restrict__ m_u,             // [D][D][3]
                        unsigned short* __restrict__ muT) {        // [3][D][D]
  int idx = blockIdx.x * blockDim.x + threadIdx.x;
  if (idx >= D_ * D_ * KU_) return;
  int i = idx % KU_;
  int d = (idx / KU_) % D_;
  int o = idx / (KU_ * D_);
  muT[((size_t)i * D_ + o) * D_ + d] = f2bf(m_u[idx]);
}

__global__ void prep_my(const float* __restrict__ m_y,             // [D][2][D]
                        unsigned short* __restrict__ myB) {        // [D][1024]
  int idx = blockIdx.x * blockDim.x + threadIdx.x;
  if (idx >= D_ * 2 * D_) return;
  myB[idx] = f2bf(m_y[idx]);
}

// Toeplitz bank: A[k][t][s] = (t>=s) ? eig_vecs[t-s][k] * eig_vals[k]^0.25 : 0
__global__ void prep_toep(const float* __restrict__ eig_vals,      // [K]
                          const float* __restrict__ eig_vecs,      // [L][K]
                          unsigned short* __restrict__ A) {        // [K][L][L]
  int idx = blockIdx.x * blockDim.x + threadIdx.x;
  if (idx >= K_ * L_ * L_) return;
  int k = idx / (L_ * L_);
  int r = idx - k * (L_ * L_);
  int t = r / L_;
  int s = r - t * L_;
  float v = 0.f;
  if (t >= s) {
    float sc = sqrtf(sqrtf(fmaxf(eig_vals[k], 0.f)));
    v = eig_vecs[(size_t)(t - s) * K_ + k] * sc;
  }
  A[idx] = f2bf(v);
}

// ------------------------- stage 1: spectral conv --------------------------
// grid (8 n-chunks of 64, 16 m-blocks of 64, b*24+k), one wave per block.
// 64x64 register-blocked tile: 16 accumulators, 16 WMMAs per K-step at
// ~32 FLOP per L2 byte. Causal K-loop skips tiles above the diagonal.
__global__ __launch_bounds__(32)
void stage1(const unsigned short* __restrict__ Atoep,   // [K][L][L]
            const unsigned short* __restrict__ uT,      // [B][D][L]
            unsigned short* __restrict__ xt) {          // [B][L][KD]
  int lane = threadIdx.x;
  int m    = lane & 15;
  int hi   = lane >> 4;
  int c0A  = hi ? 16 : 0;
  int cB   = hi ? 16 : 0;
  int d0 = blockIdx.x * 64;
  int t0 = blockIdx.y * 64;
  int z  = blockIdx.z;
  int b  = z / K_;
  int k  = z - b * K_;

  const unsigned short* Abase = Atoep + (size_t)k * L_ * L_;
  const unsigned short* Bbase = uT + (size_t)b * D_ * L_;

  v8f vz = {0.f, 0.f, 0.f, 0.f, 0.f, 0.f, 0.f, 0.f};
  v8f acc[4][4];
#pragma unroll
  for (int mi = 0; mi < 4; mi++)
#pragma unroll
    for (int ns = 0; ns < 4; ns++) acc[mi][ns] = vz;

  for (int s0 = 0; s0 < t0 + 64; s0 += 32) {           // causal: skip s > t
    v16bf bb[4];
#pragma unroll
    for (int ns = 0; ns < 4; ns++)
      bb[ns] = ldB(Bbase + (size_t)(d0 + ns * 16 + m) * L_ + s0 + cB);
#pragma unroll
    for (int mi = 0; mi < 4; mi++) {
      const unsigned short* Arow = Abase + (size_t)(t0 + mi * 16 + m) * L_;
      __builtin_prefetch(Arow + s0 + 32 + c0A, 0, 3);  // global_prefetch_b8
      v16bf a = ldA(Arow + s0 + c0A, Arow + s0 + c0A + 16);
#pragma unroll
      for (int ns = 0; ns < 4; ns++)
        acc[mi][ns] = wmma_bf16(a, bb[ns], acc[mi][ns]);
    }
  }
#pragma unroll
  for (int mi = 0; mi < 4; mi++)
#pragma unroll
    for (int ns = 0; ns < 4; ns++) {
      unsigned short* dst = xt
          + (size_t)(b * L_ + t0 + mi * 16 + hi * 8) * KD_
          + k * D_ + d0 + ns * 16 + m;
#pragma unroll
      for (int r = 0; r < 8; r++) dst[(size_t)r * KD_] = f2bf(acc[mi][ns][r]);
    }
}

// ---------------- stage 2: phi projection + fused 3-tap AR -----------------
// grid (8 n-chunks of 64, 64 m-blocks of 64), 4 waves per block.
// The phi^T B panel (4KB per K-step) is shared by all 4 waves: staged into a
// double-buffered LDS panel with global_load_async_to_lds_b128 (ASYNCcnt),
// overlapped with WMMA on the other buffer.
__global__ __launch_bounds__(128)
void stage2(const unsigned short* __restrict__ xt,     // [B*L][KD]
            const unsigned short* __restrict__ phiT,   // [D][KD]
            const unsigned short* __restrict__ u_bf,   // [B][L][D]
            const unsigned short* __restrict__ muT,    // [3][D][D]
            float* __restrict__ delta) {               // [B*L][D]
  __shared__ __align__(16) unsigned short Bpan[2][64][32];
  int tid  = threadIdx.x;
  int w    = tid >> 5;
  int lane = tid & 31;
  int m    = lane & 15;
  int hi   = lane >> 4;
  int c0A  = hi ? 16 : 0;
  int cB   = hi ? 16 : 0;
  int d0 = blockIdx.x * 64;
  int g0 = blockIdx.y * 64 + w * 16;                   // this wave's row tile
  int b  = g0 >> 10;
  int t0 = g0 & (L_ - 1);

  v8f vz = {0.f, 0.f, 0.f, 0.f, 0.f, 0.f, 0.f, 0.f};
  v8f acc[4] = {vz, vz, vz, vz};

  // thread tid stages 16B chunks tid and tid+128 of the 4KB panel (256 chunks:
  // row = c>>2 in [0,64), quarter = c&3). Two async instructions per wave.
  const unsigned short* Arow = xt + (size_t)(g0 + m) * KD_;
  int c0 = tid, c1 = tid + 128;

  // prologue: stage j0 = 0 into buffer 0
  async_ld16(&Bpan[0][c0 >> 2][(c0 & 3) * 8],
             phiT + (size_t)(d0 + (c0 >> 2)) * KD_ + 0 + (c0 & 3) * 8);
  async_ld16(&Bpan[0][c1 >> 2][(c1 & 3) * 8],
             phiT + (size_t)(d0 + (c1 >> 2)) * KD_ + 0 + (c1 & 3) * 8);

  int buf = 0;
  for (int j0 = 0; j0 < KD_; j0 += 32) {
    if (j0 + 32 < KD_) {
      int nb = buf ^ 1, jn = j0 + 32;
      async_ld16(&Bpan[nb][c0 >> 2][(c0 & 3) * 8],
                 phiT + (size_t)(d0 + (c0 >> 2)) * KD_ + jn + (c0 & 3) * 8);
      async_ld16(&Bpan[nb][c1 >> 2][(c1 & 3) * 8],
                 phiT + (size_t)(d0 + (c1 >> 2)) * KD_ + jn + (c1 & 3) * 8);
      asm volatile("s_wait_asynccnt 0x2" ::: "memory");  // current buf done
    } else {
      asm volatile("s_wait_asynccnt 0x0" ::: "memory");
    }
    __syncthreads();

    __builtin_prefetch(Arow + j0 + 32 + c0A, 0, 3);
    v16bf a = ldA(Arow + j0 + c0A, Arow + j0 + c0A + 16);
#pragma unroll
    for (int ns = 0; ns < 4; ns++) {
      v16bf bb = ldB(&Bpan[buf][ns * 16 + m][cB]);     // ds_load_b128 x2
      acc[ns] = wmma_bf16(a, bb, acc[ns]);
    }
    __syncthreads();                                   // reads done before
    buf ^= 1;                                          // buf is restaged
  }

  // fused AR: 3 shifted taps, reduction over D (small; direct global frags)
  v4u z4 = {0u, 0u, 0u, 0u};
#pragma unroll
  for (int i = 0; i < KU_; i++) {
    int t = t0 + m - i;                                // per-lane shifted row
    bool valid = (t >= 0);
    const unsigned short* Urow =
        u_bf + (size_t)(b * L_ + (valid ? t : 0)) * D_;
    for (int j0 = 0; j0 < D_; j0 += 32) {
      Frag32B f;
      if (valid) {
        f.q[0] = *(const v4u*)(Urow + j0 + c0A);
        f.q[1] = *(const v4u*)(Urow + j0 + c0A + 8);
      } else {
        f.q[0] = z4;
        f.q[1] = z4;
      }
      v16bf a = f.bf;
#pragma unroll
      for (int ns = 0; ns < 4; ns++) {
        v16bf bb = ldB(muT + ((size_t)i * D_ + d0 + ns * 16 + m) * D_ + j0 + cB);
        acc[ns] = wmma_bf16(a, bb, acc[ns]);
      }
    }
  }

#pragma unroll
  for (int ns = 0; ns < 4; ns++) {
    float* dst = delta + (size_t)(g0 + hi * 8) * D_ + d0 + ns * 16 + m;
#pragma unroll
    for (int r = 0; r < 8; r++) dst[(size_t)r * D_] = acc[ns][r];
  }
}

// ---------------------- stage 3: output recurrence -------------------------
// One workgroup, 32 waves. State S[p][row][j]: rows 0-3 = batches (rows 4-15
// stay zero), j<512 = y_{t-1} (bf16), j>=512 = y_{t-2}. Each wave owns a
// 16-wide slice of the 512 outputs; per step the K=1024 reduction runs as
// 4 interleaved WMMA chains (breaks the D->C RAW chain 4x).
__global__ __launch_bounds__(1024)
void stage3(const float* __restrict__ delta,           // [B*L][D]
            const unsigned short* __restrict__ myB,    // [D][1024]
            float* __restrict__ out) {                 // [B][L][D]
  __shared__ __align__(16) unsigned short S[2][16][1024];
  int tid = threadIdx.x;
  for (int i = tid; i < 2 * 16 * 1024; i += 1024)
    ((unsigned short*)S)[i] = 0;
  __syncthreads();

  int w    = tid >> 5;
  int lane = tid & 31;
  int m    = lane & 15;
  int hi   = lane >> 4;
  int c0A  = hi ? 16 : 0;
  int cB   = hi ? 16 : 0;
  int n0   = w * 16;

  v8f vz = {0.f, 0.f, 0.f, 0.f, 0.f, 0.f, 0.f, 0.f};
  const unsigned short* Brow = myB + (size_t)(n0 + m) * 1024;

  for (int t = 0; t < L_; t++) {
    int rd = t & 1;
    int wr = rd ^ 1;

    v8f c0 = vz, c1 = vz, c2 = vz, c3 = vz;           // c0 seeded with delta_t
    if (lane < 16) {
      c0[0] = delta[(size_t)(0 * L_ + t) * D_ + n0 + m];
      c0[1] = delta[(size_t)(1 * L_ + t) * D_ + n0 + m];
      c0[2] = delta[(size_t)(2 * L_ + t) * D_ + n0 + m];
      c0[3] = delta[(size_t)(3 * L_ + t) * D_ + n0 + m];
    }

#pragma unroll
    for (int j0 = 0; j0 < 1024; j0 += 128) {
      c0 = wmma_bf16(ldA(&S[rd][m][j0 + c0A], &S[rd][m][j0 + c0A + 16]),
                     ldB(Brow + j0 + cB), c0);
      c1 = wmma_bf16(ldA(&S[rd][m][j0 + 32 + c0A], &S[rd][m][j0 + 32 + c0A + 16]),
                     ldB(Brow + j0 + 32 + cB), c1);
      c2 = wmma_bf16(ldA(&S[rd][m][j0 + 64 + c0A], &S[rd][m][j0 + 64 + c0A + 16]),
                     ldB(Brow + j0 + 64 + cB), c2);
      c3 = wmma_bf16(ldA(&S[rd][m][j0 + 96 + c0A], &S[rd][m][j0 + 96 + c0A + 16]),
                     ldB(Brow + j0 + 96 + cB), c3);
    }
    v8f c = (c0 + c1) + (c2 + c3);

    if (lane < 16) {
#pragma unroll
      for (int r = 0; r < 4; r++) {
        float y = c[r];
        out[(size_t)(r * L_ + t) * D_ + n0 + m] = y;
        S[wr][r][n0 + m]       = f2bf(y);              // new y_{t-1}
        S[wr][r][512 + n0 + m] = S[rd][r][n0 + m];     // shift y_{t-1}->y_{t-2}
      }
    }
    __syncthreads();
  }
}

// ------------------------------ launch -------------------------------------

// workspace layout (bytes), total ~183 MB
#define OFF_A     ((size_t)0)                     // Toeplitz bf16  [K][L][L]
#define OFF_UT    ((size_t)50331648)              // u^T bf16       [B][D][L]
#define OFF_UBF   ((size_t)54525952)              // u bf16         [B][L][D]
#define OFF_PHIT  ((size_t)58720256)              // phi^T bf16     [D][KD]
#define OFF_MUT   ((size_t)71303168)              // mu^T bf16      [3][D][D]
#define OFF_MYB   ((size_t)72876032)              // my bf16        [D][1024]
#define OFF_XT    ((size_t)73924608)              // x_tilde bf16   [B*L][KD]
#define OFF_DELTA ((size_t)174587904)             // delta f32      [B*L][D]

extern "C" void kernel_launch(void* const* d_in, const int* in_sizes, int n_in,
                              void* d_out, int out_size, void* d_ws,
                              size_t ws_size, hipStream_t stream) {
  const float* inp      = (const float*)d_in[0];   // [B][L][D]
  const float* m_u      = (const float*)d_in[1];   // [D][D][3]
  const float* m_phi    = (const float*)d_in[2];   // [KD][D]
  const float* m_y      = (const float*)d_in[3];   // [D][2][D]
  const float* eig_vals = (const float*)d_in[4];   // [K]
  const float* eig_vecs = (const float*)d_in[5];   // [L][K]

  char* ws = (char*)d_ws;
  unsigned short* Atoep = (unsigned short*)(ws + OFF_A);
  unsigned short* uT    = (unsigned short*)(ws + OFF_UT);
  unsigned short* u_bf  = (unsigned short*)(ws + OFF_UBF);
  unsigned short* phiT  = (unsigned short*)(ws + OFF_PHIT);
  unsigned short* muT   = (unsigned short*)(ws + OFF_MUT);
  unsigned short* myB   = (unsigned short*)(ws + OFF_MYB);
  unsigned short* xt    = (unsigned short*)(ws + OFF_XT);
  float*          delta = (float*)(ws + OFF_DELTA);

  prep_inputs<<<(B_ * L_ * D_ + 255) / 256, 256, 0, stream>>>(inp, u_bf, uT);
  prep_phi<<<(KD_ * D_ + 255) / 256, 256, 0, stream>>>(m_phi, phiT);
  prep_mu<<<(D_ * D_ * KU_ + 255) / 256, 256, 0, stream>>>(m_u, muT);
  prep_my<<<(D_ * 2 * D_ + 255) / 256, 256, 0, stream>>>(m_y, myB);
  prep_toep<<<(K_ * L_ * L_ + 255) / 256, 256, 0, stream>>>(eig_vals, eig_vecs,
                                                            Atoep);

  stage1<<<dim3(8, 16, B_ * K_), 32, 0, stream>>>(Atoep, uT, xt);
  stage2<<<dim3(8, 64), 128, 0, stream>>>(xt, phiT, u_bf, muT, delta);
  stage3<<<1, 1024, 0, stream>>>(delta, myB, (float*)d_out);
}
